// MultiHead_44839458571007
// MI455X (gfx1250) — matrix operand order repaired and checked by reading, not compiled, not run on hardware
//
#include <hip/hip_runtime.h>
#include <hip/hip_bf16.h>

typedef __attribute__((ext_vector_type(16))) _Float16 v16h;
typedef __attribute__((ext_vector_type(8)))  _Float16 v8h;
typedef __attribute__((ext_vector_type(8)))  float    v8f;

#define DD   128
#define HH   128
#define WWD  128
#define HWW  (HH * WWD)
#define BTN  16
#define TT   8
#define DKC  32

// per-clip strides for attention weights (f16) and padded f32 scores
#define AW_CLIP  279616
#define SF_CLIP  279808

// ---- CDNA5 wave32 WMMA fragment loaders (per ISA 7.12.2 layouts) ----
// A (16-bit 16x32): lane L -> M=L%16, lh=L/16; halves h<8 -> K=8lh+h,
// h>=8 -> K=16+8lh+(h-8). From a row-major [m][k] tile this is two
// contiguous 16B runs -> 2x ds_load_b128.
__device__ __forceinline__ v16h frag_a_rm(const _Float16* p, int rs) {
    const int lane = threadIdx.x & 31, m = lane & 15, lh = lane >> 4;
    const _Float16* r = p + m * rs + 8 * lh;
    v8h lo = *(const v8h*)r;
    v8h hi = *(const v8h*)(r + 16);
    return __builtin_shufflevector(lo, hi, 0, 1, 2, 3, 4, 5, 6, 7,
                                   8, 9, 10, 11, 12, 13, 14, 15);
}
// B (16-bit 32x16): lane L -> N=L%16, lh=L/16; halves h<8 -> K=8lh+h,
// h>=8 -> K=16+8lh+(h-8). From an N-major [n][k] tile: two 16B runs.
__device__ __forceinline__ v16h frag_b_tr(const _Float16* p, int rs) {
    const int lane = threadIdx.x & 31, n = lane & 15, lh = lane >> 4;
    const _Float16* r = p + n * rs + 8 * lh;
    v8h lo = *(const v8h*)r;
    v8h hi = *(const v8h*)(r + 16);
    return __builtin_shufflevector(lo, hi, 0, 1, 2, 3, 4, 5, 6, 7,
                                   8, 9, 10, 11, 12, 13, 14, 15);
}
__device__ __forceinline__ v8f wmma_f16(v16h a, v16h b, v8f c) {
    return __builtin_amdgcn_wmma_f32_16x16x32_f16(false, a, false, b,
                                                  (short)0, c, false, false);
}

// =====================================================================
// Kernel 1: Q/K/V 1x1-conv projections (GEMM 128x128 x N), f32 -> f16 out
// grid = 3 * 16 * 256 blocks, 256 threads (8 waves)
// =====================================================================
__global__ __launch_bounds__(256) void qkv_proj_kernel(
    const float* __restrict__ x,
    const float* __restrict__ wq, const float* __restrict__ bq,
    const float* __restrict__ wk, const float* __restrict__ bk,
    const float* __restrict__ wv, const float* __restrict__ bv,
    _Float16* __restrict__ q16, _Float16* __restrict__ k16,
    _Float16* __restrict__ v16o)
{
    __shared__ alignas(16) _Float16 lw[DD * DD];   // [o][c] row-major, 32 KB
    __shared__ alignas(16) _Float16 lxT[64 * DD];  // [p][c] transposed, 16 KB

    const int tid  = threadIdx.x;
    const int wave = tid >> 5;
    const int lane = tid & 31;

    const int which = blockIdx.x % 3;
    const int n     = (blockIdx.x / 3) % BTN;
    const int p0    = ((blockIdx.x / 3) / BTN) * 64;

    const float* w    = (which == 0) ? wq : ((which == 1) ? wk : wv);
    const float* bias = (which == 0) ? bq : ((which == 1) ? bk : bv);
    _Float16*    out  = (which == 0) ? q16 : ((which == 1) ? k16 : v16o);

    for (int i = tid; i < DD * DD; i += 256) lw[i] = (_Float16)w[i];
    for (int i = tid; i < DD * 64; i += 256) {
        int c = i >> 6, p = i & 63;   // consecutive tid -> consecutive p (coalesced)
        lxT[p * DD + c] = (_Float16)x[(size_t)n * DD * HWW + (size_t)c * HWW + p0 + p];
    }
    __syncthreads();

    const int m0 = wave * 16;
    v16h afr[4];
#pragma unroll
    for (int kc = 0; kc < 4; ++kc)
        afr[kc] = frag_a_rm(&lw[m0 * DD + kc * DKC], DD);

    const int cn = lane & 15, lh = lane >> 4;
    for (int nt = 0; nt < 4; ++nt) {
        v8f acc = {};
#pragma unroll
        for (int kc = 0; kc < 4; ++kc) {
            v16h bfr = frag_b_tr(&lxT[nt * 16 * DD + kc * DKC], DD);
            acc = wmma_f16(afr[kc], bfr, acc);
        }
#pragma unroll
        for (int r = 0; r < 8; ++r) {
            int o = m0 + r + 8 * lh;
            int p = p0 + nt * 16 + cn;
            out[(size_t)n * DD * HWW + (size_t)o * HWW + p] =
                (_Float16)(acc[r] + bias[o]);
        }
    }
}

// =====================================================================
// Kernel 2a: window-attention score partials, split-K across blocks.
// Each block handles ~256 K-chunks of one (clip, scale, 16-token tile)
// and combines its partial 16 x nPad score tile with global_atomic_add_f32.
// grid = 2 * 160 blocks, 256 threads (8 waves)
// =====================================================================
__global__ __launch_bounds__(256) void attn_scores_partial_kernel(
    const _Float16* __restrict__ q16, const _Float16* __restrict__ k16,
    float* __restrict__ sF)
{
    __shared__ alignas(16) _Float16 lA[16 * DKC];   // [tok][k] row-major, 1 KB
    __shared__ alignas(16) _Float16 lB[512 * DKC];  // [j][k] N-major, 32 KB

    const int sp[4] = {0, 256, 1280, 17664};        // padded nPad^2 offsets

    const int tid  = threadIdx.x;
    const int wave = tid >> 5;
    const int lane = tid & 31;

    int bid = blockIdx.x;
    const int clip = bid / 160;  bid %= 160;
    int sc, mt, slice, nSlices;
    if      (bid < 64)  { sc = 0; mt = 0;            slice = bid;            nSlices = 64; }
    else if (bid < 96)  { sc = 1; bid -= 64;  mt = bid >> 4; slice = bid & 15; nSlices = 16; }
    else if (bid < 128) { sc = 2; bid -= 96;  mt = bid >> 2; slice = bid & 3;  nSlices = 4; }
    else                { sc = 3; bid -= 128; mt = bid;      slice = 0;        nSlices = 1; }

    const int pw   = 128 >> sc, ph = pw;
    const int ow   = 1 << sc;
    const int nTok = 8 << (2 * sc);
    const int nPad = (nTok < 16) ? 16 : nTok;
    const int nTiles = nPad >> 4;
    const float sscale = rsqrtf((float)(DKC * ph * pw));

    // K-chunk geometry: runs of runL contiguous px (rowsPerChunk rows of them)
    const int rlog  = (sc == 3) ? 4 : 5;          // log2(runL)
    const int runL  = 1 << rlog;
    const int rcl   = (sc == 0) ? 2 : ((sc == 1) ? 1 : 0);  // log2(pw/runL)
    const int lcpc  = 2 * (7 - sc) - 5;           // log2(chunks per channel)
    const int cpc   = 1 << lcpc;
    const int kcTot = DKC << lcpc;                // == 32 * ph*pw/32
    const int rowsPerChunk = 32 >> rlog;
    const int cps   = kcTot / nSlices;            // chunks per slice (== 256)
    const int k0    = slice * cps;

    v8f acc[4];
#pragma unroll
    for (int u = 0; u < 4; ++u) acc[u] = (v8f){};

    for (int kc = k0; kc < k0 + cps; ++kc) {
        const int c   = kc >> lcpc;
        const int rem = kc & (cpc - 1);
        const int py0 = (rem >> rcl) * rowsPerChunk;
        const int px0 = (rem & ((1 << rcl) - 1)) << rlog;
        const size_t chBase = (size_t)(sc * DKC + c) * HWW + (size_t)py0 * WWD + px0;

        // stage A (q): 16 tokens x 32 k, row-major
        for (int i = tid; i < 16 * DKC; i += 256) {
            int mm = i >> 5, k = i & 31;
            int tok = mt * 16 + mm;
            _Float16 val = (_Float16)0.f;
            if (tok < nTok) {
                int ttk = tok >> (2 * sc);
                int rr  = tok & ((1 << (2 * sc)) - 1);
                int oy  = rr >> sc, ox = rr & (ow - 1);
                val = q16[(size_t)(clip * TT + ttk) * DD * HWW + chBase +
                          (size_t)(oy * ph + (k >> rlog)) * WWD + ox * pw + (k & (runL - 1))];
            }
            lA[i] = val;
        }
        // stage B (k-proj): nPad tokens x 32 k, N-major [j][k]
        for (int i = tid; i < nPad * DKC; i += 256) {
            int j = i >> 5, k = i & 31;
            _Float16 val = (_Float16)0.f;
            if (j < nTok) {
                int ttk = j >> (2 * sc);
                int rr  = j & ((1 << (2 * sc)) - 1);
                int oy  = rr >> sc, ox = rr & (ow - 1);
                val = k16[(size_t)(clip * TT + ttk) * DD * HWW + chBase +
                          (size_t)(oy * ph + (k >> rlog)) * WWD + ox * pw + (k & (runL - 1))];
            }
            lB[i] = val;
        }
        __syncthreads();
        v16h afr = frag_a_rm(lA, DKC);
#pragma unroll
        for (int u = 0; u < 4; ++u) {
            int jt = wave + u * 8;
            if (jt < nTiles) {
                v16h bfr = frag_b_tr(&lB[jt * 16 * DKC], DKC);
                acc[u] = wmma_f16(afr, bfr, acc[u]);
            }
        }
        __syncthreads();
    }

    // combine scaled partial scores via f32 atomics
    {
        float* dst = sF + (size_t)clip * SF_CLIP + sp[sc];
        const int nn = lane & 15, lh = lane >> 4;
#pragma unroll
        for (int u = 0; u < 4; ++u) {
            int jt = wave + u * 8;
            if (jt < nTiles) {
#pragma unroll
                for (int r = 0; r < 8; ++r)
                    atomicAdd(&dst[(size_t)(mt * 16 + r + 8 * lh) * nPad + jt * 16 + nn],
                              acc[u][r] * sscale);
            }
        }
    }
}

// =====================================================================
// Kernel 2b: softmax over combined scores -> f16 attention weights
// grid = 2*43 blocks (clip, scale, 16-token tile)
// =====================================================================
__global__ __launch_bounds__(64) void attn_softmax_kernel(
    const float* __restrict__ sF, _Float16* __restrict__ attnW)
{
    const int sp[4]   = {0, 256, 1280, 17664};
    const int soff[4] = {0, 64, 1088, 17472};

    const int tid = threadIdx.x;
    int bid = blockIdx.x;
    const int clip = bid / 43;  bid %= 43;
    int sc, mt;
    if      (bid < 1)  { sc = 0; mt = bid; }
    else if (bid < 3)  { sc = 1; mt = bid - 1; }
    else if (bid < 11) { sc = 2; mt = bid - 3; }
    else               { sc = 3; mt = bid - 11; }

    const int nTok = 8 << (2 * sc);
    const int nPad = (nTok < 16) ? 16 : nTok;

    if (tid < 16) {
        int m = mt * 16 + tid;
        if (m < nTok) {
            const float* row = sF + (size_t)clip * SF_CLIP + sp[sc] + (size_t)m * nPad;
            float mx = -3.0e38f;
            for (int j = 0; j < nTok; ++j) mx = fmaxf(mx, row[j]);
            float sm = 0.f;
            for (int j = 0; j < nTok; ++j) sm += __expf(row[j] - mx);
            float inv = 1.f / sm;
            _Float16* dst = attnW + (size_t)clip * AW_CLIP + soff[sc] + (size_t)m * nTok;
            for (int j = 0; j < nTok; ++j)
                dst[j] = (_Float16)(__expf(row[j] - mx) * inv);
        }
    }
}

// =====================================================================
// Kernel 3: y = attn @ V, one 16x16 WMMA tile per wave
// grid = 20480 blocks x 256 threads (163840 wave-tiles)
// =====================================================================
__global__ __launch_bounds__(256) void attn_apply_kernel(
    const _Float16* __restrict__ v16g, const _Float16* __restrict__ attnW,
    _Float16* __restrict__ att16)
{
    const int soff[4] = {0, 64, 1088, 17472};
    const int tid = threadIdx.x, wave = tid >> 5, lane = tid & 31;
    int gt = blockIdx.x * 8 + wave;
    const int clip = gt / 81920;
    int r = gt % 81920;
    int sc, mt, ft;
    if      (r < 32768) { sc = 0; mt = 0;            ft = r; }
    else if (r < 49152) { sc = 1; r -= 32768; mt = r >> 13; ft = r & 8191; }
    else if (r < 65536) { sc = 2; r -= 49152; mt = r >> 11; ft = r & 2047; }
    else                { sc = 3; r -= 65536; mt = r >> 9;  ft = r & 511; }

    const int pw   = 128 >> sc, ph = pw;
    const int ow   = 1 << sc;
    const int nTok = 8 << (2 * sc);
    const int npxc = pw >> 4;
    const int csh  = 10 - 2 * sc;           // log2(ph*npxc)
    const int c    = ft >> csh;
    int rem        = ft & ((1 << csh) - 1);
    const int pyl  = 3 - sc;                // log2(npxc)
    const int py   = rem >> pyl;
    const int pxc  = rem & (npxc - 1);
    const int x0   = pxc * 16;

    const _Float16* aw = attnW + (size_t)clip * AW_CLIP + soff[sc];

    const int m  = lane & 15;
    const int lh = lane >> 4;
    const int nChunks = (nTok + 31) >> 5;
    const int mtok = mt * 16 + m;

    v8f acc = {};
    for (int jc = 0; jc < nChunks; ++jc) {
        v16h afr, bfr;
        if (sc > 0) {
            // rows contiguous, all bounds exact: two 16B vector loads
            const _Float16* ap = aw + (size_t)mtok * nTok + jc * 32 + 8 * lh;
            v8h lo = *(const v8h*)ap;
            v8h hi = *(const v8h*)(ap + 16);
            afr = __builtin_shufflevector(lo, hi, 0, 1, 2, 3, 4, 5, 6, 7,
                                          8, 9, 10, 11, 12, 13, 14, 15);
        } else {
#pragma unroll
            for (int h = 0; h < 16; ++h) {
                int kk = ((h & 8) << 1) + (lh << 3) + (h & 7);
                _Float16 av = (_Float16)0.f;
                if (mtok < nTok && kk < nTok) av = aw[(size_t)mtok * nTok + kk];
                afr[h] = av;
            }
        }
#pragma unroll
        for (int h = 0; h < 16; ++h) {
            int j = jc * 32 + ((h & 8) << 1) + (lh << 3) + (h & 7);
            _Float16 bv = (_Float16)0.f;
            if (j < nTok) {
                int ttk = j >> (2 * sc);
                int rr  = j & ((1 << (2 * sc)) - 1);
                int oy  = rr >> sc, ox = rr & (ow - 1);
                bv = v16g[((size_t)(clip * TT + ttk) * DD + (sc * DKC + c)) * HWW +
                          (size_t)(oy * ph + py) * WWD + (ox * pw + x0 + m)];
            }
            bfr[h] = bv;
        }
        acc = wmma_f16(afr, bfr, acc);
    }

#pragma unroll
    for (int rI = 0; rI < 8; ++rI) {
        int mo = mt * 16 + rI + 8 * lh;
        if (mo < nTok) {
            int ttk = mo >> (2 * sc);
            int rr  = mo & ((1 << (2 * sc)) - 1);
            int oy  = rr >> sc, ox = rr & (ow - 1);
            att16[((size_t)(clip * TT + ttk) * DD + (sc * DKC + c)) * HWW +
                  (size_t)(oy * ph + py) * WWD + (ox * pw + x0 + m)] = (_Float16)acc[rI];
        }
    }
}

// =====================================================================
// Kernel 4: 3x3 conv (implicit GEMM) + bias + ReLU, f16 in / f32 out
// grid = 16*128 blocks (one image row each), 256 threads (8 waves)
// =====================================================================
__global__ __launch_bounds__(256) void conv3_kernel(
    const _Float16* __restrict__ att16, const float* __restrict__ wo,
    const float* __restrict__ bo, float* __restrict__ out)
{
    __shared__ alignas(16) _Float16 lwA[3 * 128 * 32]; // [kx][o][ci] row-major, 24 KB
    __shared__ alignas(16) _Float16 lrowT[132 * 32];   // [1+x][ci] transposed, ~8.3 KB

    const int tid = threadIdx.x, wave = tid >> 5, lane = tid & 31;
    const int n = blockIdx.x >> 7;
    const int y = blockIdx.x & 127;

    v8f acc[8];
#pragma unroll
    for (int xt = 0; xt < 8; ++xt) acc[xt] = (v8f){};

    const int o0 = wave * 16;
    const int mI = lane & 15, lh = lane >> 4;

    for (int ky = 0; ky < 3; ++ky) {
        const int yy = y + ky - 1;
        const bool yok = (yy >= 0) && (yy < HH);
        for (int cc = 0; cc < 4; ++cc) {
            for (int i = tid; i < 128 * 32 * 3; i += 256) {
                int o = i / 96, rr = i % 96, ci = rr / 3, kx = rr % 3;
                lwA[(kx * 128 + o) * 32 + ci] =
                    (_Float16)wo[(((size_t)o * DD + cc * 32 + ci) * 3 + ky) * 3 + kx];
            }
            for (int i = tid; i < 32 * 130; i += 256) {
                int ci = i / 130, xx = i % 130, xg = xx - 1;
                _Float16 val = (_Float16)0.f;
                if (yok && xg >= 0 && xg < WWD)
                    val = att16[((size_t)n * DD + cc * 32 + ci) * HWW +
                                (size_t)yy * WWD + xg];
                lrowT[xx * 32 + ci] = val;
            }
            __syncthreads();
            for (int kx = 0; kx < 3; ++kx) {
                v16h afr = frag_a_rm(&lwA[(kx * 128 + o0) * 32], 32);
#pragma unroll
                for (int xt = 0; xt < 8; ++xt) {
                    v16h bfr = frag_b_tr(&lrowT[(xt * 16 + kx) * 32], 32);
                    acc[xt] = wmma_f16(afr, bfr, acc[xt]);
                }
            }
            __syncthreads();
        }
    }

#pragma unroll
    for (int xt = 0; xt < 8; ++xt) {
#pragma unroll
        for (int rI = 0; rI < 8; ++rI) {
            int o = o0 + rI + 8 * lh;
            float v = acc[xt][rI] + bo[o];
            out[((size_t)n * DD + o) * HWW + (size_t)y * WWD + xt * 16 + mI] =
                fmaxf(v, 0.f);
        }
    }
}

// =====================================================================
extern "C" void kernel_launch(void* const* d_in, const int* in_sizes, int n_in,
                              void* d_out, int out_size, void* d_ws, size_t ws_size,
                              hipStream_t stream) {
    const float* x  = (const float*)d_in[0];
    const float* wq = (const float*)d_in[1];
    const float* bq = (const float*)d_in[2];
    const float* wk = (const float*)d_in[3];
    const float* bk = (const float*)d_in[4];
    const float* wv = (const float*)d_in[5];
    const float* bv = (const float*)d_in[6];
    const float* wo = (const float*)d_in[7];
    const float* bo = (const float*)d_in[8];
    float* out = (float*)d_out;

    char* ws = (char*)d_ws;
    const size_t planeBytes = (size_t)BTN * DD * HWW * sizeof(_Float16); // 64 MB
    _Float16* q16   = (_Float16*)(ws);
    _Float16* k16   = (_Float16*)(ws + planeBytes);
    _Float16* v16g  = (_Float16*)(ws + 2 * planeBytes);
    _Float16* a16   = (_Float16*)(ws + 3 * planeBytes);
    _Float16* attnW = (_Float16*)(ws + 4 * planeBytes);
    float*    sF    = (float*)(ws + 4 * planeBytes + (size_t)2 * 1024 * 1024);

    // zero the f32 score accumulation buffer (graph-capture-legal memset)
    hipMemsetAsync(sF, 0, (size_t)2 * SF_CLIP * sizeof(float), stream);

    hipLaunchKernelGGL(qkv_proj_kernel, dim3(3 * BTN * 256), dim3(256), 0, stream,
                       x, wq, bq, wk, bk, wv, bv, q16, k16, v16g);
    hipLaunchKernelGGL(attn_scores_partial_kernel, dim3(320), dim3(256), 0, stream,
                       q16, k16, sF);
    hipLaunchKernelGGL(attn_softmax_kernel, dim3(86), dim3(64), 0, stream,
                       sF, attnW);
    hipLaunchKernelGGL(attn_apply_kernel, dim3(20480), dim3(256), 0, stream,
                       v16g, attnW, a16);
    hipLaunchKernelGGL(conv3_kernel, dim3(BTN * HH), dim3(256), 0, stream,
                       a16, wo, bo, out);
}
